// DynamicModel_59210419142658
// MI455X (gfx1250) — matrix operand order repaired and checked by reading, not compiled, not run on hardware
//
#include <hip/hip_runtime.h>
#include <hip/hip_bf16.h>
#include <cstdint>
#include <cstddef>

// ---------------------------------------------------------------------------
// Tsit5 neural-ODE rollout, fused persistent kernel for gfx1250 (MI455X).
//   B=4096, S=50, OBS=32, ACT=8, H=512.
//   One workgroup = 64 batch rows, 512 threads (16 waves, 4M x 4N wave grid);
//   all rollout state lives in LDS.  GEMMs: v_wmma_f32_16x16x32_bf16.
//   Weight K-panels: TDM (tensor_load_to_lds), triple-buffered in LDS with a
//   single barrier per K-chunk; B-fragments double-buffered in VGPRs.
//   16 waves -> 64 acc VGPRs/wave: no vgpr_msb pressure, 4 waves/SIMD.
// ---------------------------------------------------------------------------

typedef __attribute__((ext_vector_type(16))) __bf16        v16bf;
typedef __attribute__((ext_vector_type(8)))  float         v8f;
typedef __attribute__((ext_vector_type(4)))  unsigned int  v4u;
typedef __attribute__((ext_vector_type(8)))  int           v8i;
typedef __attribute__((ext_vector_type(4)))  int           v4i;

#define MT      64      // batch rows per workgroup
#define THREADS 512     // 16 waves of 32
#define NBLK    64      // 4096 / MT
#define SEQ     50
#define OBS     32
#define ACTD    8
#define HDIM    512
#define KPAD    64      // K of layer 1 padded 40 -> 64

// padded LDS strides (elements) to dodge bank conflicts on fragment loads
#define HSTR 520        // h tiles      (row = 1040B -> +4 banks/row)
#define XSTR 72         // x tile       (row = 144B  -> +36 banks/row)
#define PSTR 40         // weight panel (row = 80B   -> +20 banks/row, TDM pad)

// LDS byte offsets (total 322560 B <= 320KB / WGP)
#define OFF_HA   0
#define OFF_HB   66560                  // 64*520*2
#define OFF_P0   133120                 // 3 x (512*40*2 = 40960) panels
#define OFF_P1   174080
#define OFF_P2   215040
#define OFF_X    256000                 // 64*72*2 = 9216
#define OFF_Y    265216                 // 64*32*4 = 8192
#define OFF_K    273408                 // 6 x 8192 k-buffers
#define LDS_BYTES 322560

#define DTF 0.05f
__constant__ float AC[6][5] = {
  {0.f,0.f,0.f,0.f,0.f},
  {0.161f,0.f,0.f,0.f,0.f},
  {-0.008480655492356989f, 0.335480655492357f, 0.f,0.f,0.f},
  {2.8971530571054935f, -6.359448489975075f, 4.3622954328695815f, 0.f,0.f},
  {5.325864828439257f, -11.748883564062828f, 7.4955393428898365f, -0.09249506636175525f, 0.f},
  {5.86145544294642f, -12.92096931784711f, 8.159367898576159f, -0.071584973281401f, -0.028269050394068383f},
};
#define BW1 0.09646076681806523f
#define BW2 0.01f
#define BW3 0.4798896504144996f
#define BW4 1.379008574103742f
#define BW5 (-3.290069515436081f)
#define BW6 2.324710524099774f

// ---------------------------------------------------------------------------
__device__ __forceinline__ v8f wmma_bf16(v16bf a, v16bf b, v8f c) {
  return __builtin_amdgcn_wmma_f32_16x16x32_bf16(false, a, false, b,
                                                 (short)0, c, false, false);
}

__device__ __forceinline__ float softplus(float v) {
  return fmaxf(v, 0.f) + log1pf(__expf(-fabsf(v)));
}

// ---------------------------------------------------------------------------
// TDM: DMA one (NDIM x 32)-element bf16 K-panel of the (NDIM x KDIM) weight
// matrix into LDS at lds_off. D# per CDNA5 ISA:
//  - 2D tensor, data_size=2B, tile_dim0=32 (K), tile_dim1=NDIM (rows)
//  - pad_enable: after every 16 DWORDs (=one 64B row) insert 4 DWORDs (16B)
//    -> LDS row stride 80B = PSTR elements (bank-conflict-free fragments).
// ---------------------------------------------------------------------------
template <int KDIM, int NDIM>
__device__ __forceinline__ void tdm_issue(unsigned lds_off,
                                          const __bf16* __restrict__ Wt, int kc) {
  unsigned long long ga = (unsigned long long)(uintptr_t)Wt
                        + (unsigned long long)(kc * 64);        // 32 elem * 2B
  v4u g0 = { 1u,                                   // count=1, user mode
             lds_off,                              // lds_addr
             (unsigned)ga,                         // global_addr[31:0]
             (unsigned)((ga >> 32) & 0x01FFFFFFull) | (2u << 30) }; // type=2
  v8i g1 = { (1 << 16) | (1 << 20) | (3 << 22) | (3 << 25),
             //  ^data_size=2B  ^pad_en  ^pad_interval=16DW  ^pad_amount=4DW
             (int)((KDIM & 0xFFFF) << 16),         // tensor_dim0 lo16
             (int)((NDIM & 0xFFFF) << 16),         // tensor_dim1 lo16
             (int)(32u << 16),                     // tile_dim0 = 32
             NDIM,                                 // tile_dim1 = NDIM
             KDIM,                                 // tensor_dim0_stride
             0, 0 };
  v4i z4 = {0, 0, 0, 0};
#if defined(__clang_major__) && (__clang_major__ >= 23)
  v8i z8 = {0, 0, 0, 0, 0, 0, 0, 0};
  __builtin_amdgcn_tensor_load_to_lds(g0, g1, z4, z4, z8, 0);
#else
  __builtin_amdgcn_tensor_load_to_lds(g0, g1, z4, z4, 0);
#endif
}

// ---------------------------------------------------------------------------
// One dense layer: dest = act( A[MT x KDIM] @ Wt^T + b ).  Wt (NDIM x KDIM)
// bf16 row-major (pre-transposed).  TDM triple-buffers K-panels in LDS:
// one barrier per K-chunk; the panel overwritten for kc+2 is the one last
// read at kc-1, already protected by this iteration's barrier.
// ---------------------------------------------------------------------------
template <int KDIM, int NDIM, bool ACT, bool DESTF32>
__device__ __forceinline__ void dense(const __bf16* __restrict__ A, int astride,
                                      const __bf16* __restrict__ Wt,
                                      const float* __restrict__ bias,
                                      __bf16* destH, float* destF,
                                      char* smem, int tid) {
  constexpr int NK  = KDIM / 32;
  constexpr int NCH = (NDIM == HDIM) ? 8 : 1;    // 16-col chunks per wave

  const int wave = tid >> 5, lane = tid & 31;
  const int rowbase = 16 * (wave & 3);           // 4 M-groups
  const int ngrp = wave >> 2;                    // 4 N-groups
  const int ncolbase = (NDIM == HDIM) ? (ngrp * 128) : (ngrp * 16);
  const bool active = (NDIM == HDIM) || (ngrp < 2);   // N=32: waves 8-15 idle
  const bool issuer = (tid < 32);                // wave 0 drives the TDM

  const __bf16* pa = (const __bf16*)(smem + OFF_P0);   // panel holding kc
  const __bf16* pb = (const __bf16*)(smem + OFF_P1);   // panel holding kc+1
  const __bf16* pc = (const __bf16*)(smem + OFF_P2);   // target for kc+2
  unsigned oa = OFF_P0, ob = OFF_P1, oc = OFF_P2;

  __syncthreads();              // previous dense done reading the panels
  if (issuer) {
    tdm_issue<KDIM, NDIM>(oa, Wt, 0);
    if constexpr (NK > 1) tdm_issue<KDIM, NDIM>(ob, Wt, 1);
  }

  v8f acc[NCH];
#pragma unroll
  for (int c = 0; c < NCH; ++c)
#pragma unroll
    for (int r = 0; r < 8; ++r) acc[c][r] = 0.f;

  // hoisted per-lane fragment bases; chunk steps fold into ds_load offsets
  const __bf16* arow = A + (rowbase + (lane & 15)) * astride + ((lane >> 4) << 3);

  // B-fragments double-buffered in VGPRs: load chunk c+1 before WMMA c.
  auto compute = [&](int kc, const __bf16* pan) {
    if (!active) return;
    union U { v16bf v; uint4 q[2]; } ua, ub0, ub1;
    const __bf16* ap = arow + kc * 32;
    ua.q[0] = *(const uint4*)ap;                   // A: K {0..7 | 8..15}
    ua.q[1] = *(const uint4*)(ap + 16);            // A: K {16..23 | 24..31}
    const __bf16* bp = pan + (ncolbase + (lane & 15)) * PSTR + ((lane >> 4) << 3);
    ub0.q[0] = *(const uint4*)bp;
    ub0.q[1] = *(const uint4*)(bp + 16);
#pragma unroll
    for (int c = 0; c < NCH; c += 2) {
      if (c + 1 < NCH) {
        ub1.q[0] = *(const uint4*)(bp + (c + 1) * 16 * PSTR);
        ub1.q[1] = *(const uint4*)(bp + (c + 1) * 16 * PSTR + 16);
      }
      acc[c] = wmma_bf16(ua.v, ub0.v, acc[c]);
      if (c + 2 < NCH) {
        ub0.q[0] = *(const uint4*)(bp + (c + 2) * 16 * PSTR);
        ub0.q[1] = *(const uint4*)(bp + (c + 2) * 16 * PSTR + 16);
      }
      if (c + 1 < NCH) acc[c + 1] = wmma_bf16(ua.v, ub1.v, acc[c + 1]);
    }
  };

  for (int kc = 0; kc < NK - 1; ++kc) {
    if (issuer) __builtin_amdgcn_s_wait_tensorcnt(1);  // panel kc landed
    __syncthreads();              // also: panel (kc+2)%3 free (read at kc-1)
    if (issuer && (kc + 2 < NK)) tdm_issue<KDIM, NDIM>(oc, Wt, kc + 2);
    compute(kc, pa);
    const __bf16* tp = pa; pa = pb; pb = pc; pc = tp;   // rotate panels
    unsigned to = oa; oa = ob; ob = oc; oc = to;
  }
  if (issuer) __builtin_amdgcn_s_wait_tensorcnt(0);
  __syncthreads();
  compute(NK - 1, pa);

  // epilogue: C/D layout -> VGPR r, lane l : row = r + 8*(l>=16), col = l&15
  if (active) {
#pragma unroll
    for (int c = 0; c < NCH; ++c) {
      int ncol = ncolbase + 16 * c + (lane & 15);
      float b = bias[ncol];                        // L2-resident, tiny
#pragma unroll
      for (int r = 0; r < 8; ++r) {
        int row = rowbase + r + ((lane >> 4) << 3);
        float v = acc[c][r] + b;
        if (ACT) v = softplus(v);
        if (DESTF32) destF[row * OBS + ncol] = v;
        else         destH[row * HSTR + ncol] = (__bf16)v;
      }
    }
  }
}

// ---------------------------------------------------------------------------
// One MLP evaluation f(x) -> kdst.  noinline: single code copy for the 300
// sequential evaluations (I-cache footprint).
// ---------------------------------------------------------------------------
__device__ __noinline__ void feval(char* smem,
                                   const __bf16* __restrict__ w1t,
                                   const __bf16* __restrict__ w2t,
                                   const __bf16* __restrict__ w3t,
                                   const __bf16* __restrict__ w4t,
                                   const float* __restrict__ b1,
                                   const float* __restrict__ b2,
                                   const float* __restrict__ b3,
                                   const float* __restrict__ b4,
                                   float* kdst, int tid) {
  __bf16* hA = (__bf16*)(smem + OFF_HA);
  __bf16* hB = (__bf16*)(smem + OFF_HB);
  __bf16* x  = (__bf16*)(smem + OFF_X);
  dense<KPAD, HDIM, true,  false>(x,  XSTR, w1t, b1, hA, nullptr, smem, tid);
  dense<HDIM, HDIM, true,  false>(hA, HSTR, w2t, b2, hB, nullptr, smem, tid);
  dense<HDIM, HDIM, true,  false>(hB, HSTR, w3t, b3, hA, nullptr, smem, tid);
  dense<HDIM, OBS,  false, true >(hA, HSTR, w4t, b4, nullptr, kdst, smem, tid);
}

// ---------------------------------------------------------------------------
// Stage input: x[:, :32] = y + DT * sum_j AC[s][j] * k_j ; x[:, 32:40] = action
// ---------------------------------------------------------------------------
template <int STG>
__device__ __forceinline__ void build_x(__bf16* x, const float* y, const float* kb,
                                        const float* __restrict__ act,
                                        int row0, int aidx, int tid) {
  __syncthreads();
  for (int e = tid; e < MT * OBS; e += THREADS) {
    int r = e >> 5, c = e & 31;
    float v = y[e];
#pragma unroll
    for (int j = 0; j < STG - 1; ++j)
      v += DTF * AC[STG - 1][j] * kb[j * (MT * OBS) + e];
    x[r * XSTR + c] = (__bf16)v;
  }
  for (int e = tid; e < MT * ACTD; e += THREADS) {
    int r = e >> 3, c = e & 7;
    x[r * XSTR + OBS + c] =
        (__bf16)act[(size_t)(row0 + r) * (SEQ * ACTD) + aidx * ACTD + c];
  }
}

// ---------------------------------------------------------------------------
__global__ __launch_bounds__(THREADS, 1)
void tsit5_rollout(const float* __restrict__ init_obs,
                   const float* __restrict__ act,
                   const float* __restrict__ b1, const float* __restrict__ b2,
                   const float* __restrict__ b3, const float* __restrict__ b4,
                   const __bf16* __restrict__ w1t, const __bf16* __restrict__ w2t,
                   const __bf16* __restrict__ w3t, const __bf16* __restrict__ w4t,
                   float* __restrict__ out) {
  extern __shared__ char smem[];
  __bf16* x  = (__bf16*)(smem + OFF_X);
  float*  y  = (float*)(smem + OFF_Y);
  float*  kb = (float*)(smem + OFF_K);     // 6 buffers of MT*OBS floats

  const int tid  = threadIdx.x;
  const int row0 = blockIdx.x * MT;

  // init: y tile, zero x's K-pad columns [40,64)
  for (int e = tid; e < MT * OBS; e += THREADS) y[e] = init_obs[(size_t)row0 * OBS + e];
  for (int e = tid; e < MT * (KPAD - OBS - ACTD); e += THREADS) {
    int r = e / (KPAD - OBS - ACTD), c = e % (KPAD - OBS - ACTD);
    x[r * XSTR + OBS + ACTD + c] = (__bf16)0.f;
  }

  for (int n = 0; n < SEQ; ++n) {
    const int a6 = (n + 1 < SEQ) ? n + 1 : SEQ - 1;   // c6 = 1.0 -> next action
    build_x<1>(x, y, kb, act, row0, n,  tid);
    feval(smem, w1t, w2t, w3t, w4t, b1, b2, b3, b4, kb + 0 * (MT * OBS), tid);
    build_x<2>(x, y, kb, act, row0, n,  tid);
    feval(smem, w1t, w2t, w3t, w4t, b1, b2, b3, b4, kb + 1 * (MT * OBS), tid);
    build_x<3>(x, y, kb, act, row0, n,  tid);
    feval(smem, w1t, w2t, w3t, w4t, b1, b2, b3, b4, kb + 2 * (MT * OBS), tid);
    build_x<4>(x, y, kb, act, row0, n,  tid);
    feval(smem, w1t, w2t, w3t, w4t, b1, b2, b3, b4, kb + 3 * (MT * OBS), tid);
    build_x<5>(x, y, kb, act, row0, n,  tid);
    feval(smem, w1t, w2t, w3t, w4t, b1, b2, b3, b4, kb + 4 * (MT * OBS), tid);
    build_x<6>(x, y, kb, act, row0, a6, tid);
    feval(smem, w1t, w2t, w3t, w4t, b1, b2, b3, b4, kb + 5 * (MT * OBS), tid);

    __syncthreads();                                   // k6 visible
    for (int e = tid; e < MT * OBS; e += THREADS) {
      int r = e >> 5, c = e & 31;
      float v = y[e] + DTF * (BW1 * kb[e] +
                              BW2 * kb[1 * MT * OBS + e] +
                              BW3 * kb[2 * MT * OBS + e] +
                              BW4 * kb[3 * MT * OBS + e] +
                              BW5 * kb[4 * MT * OBS + e] +
                              BW6 * kb[5 * MT * OBS + e]);
      y[e] = v;
      out[(size_t)(row0 + r) * (SEQ * OBS) + n * OBS + c] = v;
    }
  }
}

// ---------------------------------------------------------------------------
// Prep: transpose f32 weights (K x N) -> bf16 (N x Kpad), zero-padding K.
// ---------------------------------------------------------------------------
__global__ void prep_transpose_bf16(const float* __restrict__ src,
                                    __bf16* __restrict__ dst,
                                    int K, int N, int Kpad) {
  int i = blockIdx.x * blockDim.x + threadIdx.x;
  if (i >= N * Kpad) return;
  int n = i / Kpad, k = i - n * Kpad;
  float v = (k < K) ? src[(size_t)k * N + n] : 0.f;
  dst[i] = (__bf16)v;
}

// ---------------------------------------------------------------------------
extern "C" void kernel_launch(void* const* d_in, const int* in_sizes, int n_in,
                              void* d_out, int out_size, void* d_ws, size_t ws_size,
                              hipStream_t stream) {
  (void)in_sizes; (void)n_in; (void)out_size; (void)ws_size;
  const float* init = (const float*)d_in[0];
  const float* act  = (const float*)d_in[1];
  const float* W1 = (const float*)d_in[2]; const float* b1 = (const float*)d_in[3];
  const float* W2 = (const float*)d_in[4]; const float* b2 = (const float*)d_in[5];
  const float* W3 = (const float*)d_in[6]; const float* b3 = (const float*)d_in[7];
  const float* W4 = (const float*)d_in[8]; const float* b4 = (const float*)d_in[9];

  char* ws = (char*)d_ws;
  __bf16* w1t = (__bf16*)(ws);                       // 512 x 64
  __bf16* w2t = (__bf16*)(ws + 65536);               // 512 x 512
  __bf16* w3t = (__bf16*)(ws + 65536 + 524288);      // 512 x 512
  __bf16* w4t = (__bf16*)(ws + 65536 + 2 * 524288);  // 32  x 512

  prep_transpose_bf16<<<(512 * KPAD + 255) / 256, 256, 0, stream>>>(W1, w1t, OBS + ACTD, HDIM, KPAD);
  prep_transpose_bf16<<<(512 * 512 + 255) / 256, 256, 0, stream>>>(W2, w2t, HDIM, HDIM, HDIM);
  prep_transpose_bf16<<<(512 * 512 + 255) / 256, 256, 0, stream>>>(W3, w3t, HDIM, HDIM, HDIM);
  prep_transpose_bf16<<<(32 * 512 + 255) / 256, 256, 0, stream>>>(W4, w4t, HDIM, OBS, HDIM);

  (void)hipFuncSetAttribute((const void*)tsit5_rollout,
                            hipFuncAttributeMaxDynamicSharedMemorySize, LDS_BYTES);
  tsit5_rollout<<<NBLK, THREADS, LDS_BYTES, stream>>>(
      init, act, b1, b2, b3, b4, w1t, w2t, w3t, w4t, (float*)d_out);
}
